// WindowAttentionLayer_1116691497036
// MI455X (gfx1250) — matrix operand order repaired and checked by reading, not compiled, not run on hardware
//
#include <hip/hip_runtime.h>

// ---------------- types ----------------
typedef __bf16 bf16_t;
typedef __attribute__((ext_vector_type(4)))  float   v4f;
typedef __attribute__((ext_vector_type(8)))  float   v8f;
typedef __attribute__((ext_vector_type(8)))  __bf16  v8bf;
typedef __attribute__((ext_vector_type(16))) __bf16  v16bf;

// ---------------- problem constants ----------------
#define NWIN   4096
#define NTOK   64
#define CDIM   512
#define HEADS  16
#define DHEAD  32

// LDS strides (elements); all fragment bases stay 16B aligned
#define XS   520   // x / O buffers: 64 x 520 bf16 (token-major)
#define QKS  40    // q,k buffers:   64 x 40  bf16 (token-major)
#define VTS  72    // v transposed:  32 x 72  bf16 (dh-major)
#define SCS  68    // scoresT:       64 x 68  f32  (key-major)
#define WBS  72    // softmax bf16:  64 x 72  bf16 (query-major)

// per-group LDS spans (elements)
#define GRP_BF   12032   // q(2560)+k(2560)+vT(2304)+wbf(4608)
#define GRP_SC   4352    // 64*68 f32

// ws layout (bytes)
#define WS_QKVW_OFF  0u
#define WS_PROJW_OFF 1572864u   // 786432 bf16
#define WS_BIAS_OFF  2097152u   // + 524288 (proj) ; bias = 262144 f32

// ---------------- fragment helpers ----------------
__device__ __forceinline__ v16bf cat16(v8bf lo, v8bf hi) {
  v16bf r;
#pragma unroll
  for (int i = 0; i < 8; ++i) { r[i] = lo[i]; r[i + 8] = hi[i]; }
  return r;
}

// A fragment (16x32 bf16) from a row-major [M][K] LDS matrix.
// Lane l: row M=r0+(l&15); elems 0..7 = K k0+(l>>4)*8+{0..7}, elems 8..15 = +16.
__device__ __forceinline__ v16bf load_A(const bf16_t* M, int stride, int r0, int k0, int lane) {
  const bf16_t* p = M + (r0 + (lane & 15)) * stride + k0 + ((lane >> 4) << 3);
  v8bf lo = *(const v8bf*)p;
  v8bf hi = *(const v8bf*)(p + 16);
  return cat16(lo, hi);
}

// B fragment (32x16 bf16) from an [N][K]-stored matrix.
// Lane l: col N=n0+(l&15); elems 0..7 = K k0+(l>>4)*8+{0..7}, elems 8..15 = +16.
__device__ __forceinline__ v16bf load_B(const bf16_t* Bt, int stride, int n0, int k0, int lane) {
  const bf16_t* p = Bt + (n0 + (lane & 15)) * stride + k0 + ((lane >> 4) << 3);
  v8bf lo = *(const v8bf*)p;
  v8bf hi = *(const v8bf*)(p + 16);
  return cat16(lo, hi);
}

// Pre-swizzled global weight block (1KB = one 16x32 fragment): lane's 16 elems contiguous.
__device__ __forceinline__ v16bf load_W(const bf16_t* blk, int lane) {
  const bf16_t* p = blk + lane * 16;
  v8bf lo = *(const v8bf*)p;
  v8bf hi = *(const v8bf*)(p + 8);
  return cat16(lo, hi);
}

__device__ __forceinline__ v8f wmma_bf16(v16bf a, v16bf b, v8f c) {
  return __builtin_amdgcn_wmma_f32_16x16x32_bf16(false, a, false, b, (short)0, c, false, false);
}

// ---------------- prep: weights -> bf16 A-fragment swizzle, bias pre-gather ----------------
// A-layout block (ktile, mtile): lane l holds row n = mtile*16+(l&15);
// elem i -> k = ktile*32 + (l>>4)*8 + (i&7) + (i>=8 ? 16 : 0).
__global__ __launch_bounds__(256) void prep_kernel(
    const float* __restrict__ qkv_w, const float* __restrict__ proj_w,
    const int* __restrict__ rel_index, const float* __restrict__ table,
    bf16_t* __restrict__ wqkv, bf16_t* __restrict__ wproj, float* __restrict__ biasws)
{
  int tid = blockIdx.x * 256 + threadIdx.x;
  if (tid < 786432) {                       // qkv_w^T: 16 ktiles x 96 mtiles
    int blk = tid >> 9, r = tid & 511;
    int lane = r >> 4, i = r & 15;
    int ktile = blk / 96, mtile = blk % 96;
    int k = ktile * 32 + ((lane >> 4) << 3) + (i & 7) + ((i >> 3) << 4);
    int n = mtile * 16 + (lane & 15);
    wqkv[tid] = (bf16_t)qkv_w[k * 1536 + n];
  }
  if (tid < 262144) {                       // proj_w^T: 16 ktiles x 32 mtiles
    int blk = tid >> 9, r = tid & 511;
    int lane = r >> 4, i = r & 15;
    int ktile = blk / 32, mtile = blk % 32;
    int k = ktile * 32 + ((lane >> 4) << 3) + (i & 7) + ((i >> 3) << 4);
    int n = mtile * 16 + (lane & 15);
    wproj[tid] = (bf16_t)proj_w[k * 512 + n];
  }
  if (tid < 65536) {                        // biasT [H][key m][query n]
    int h = tid >> 12;
    int m = (tid >> 6) & 63;
    int n = tid & 63;
    biasws[tid] = table[rel_index[n * 64 + m] * 16 + h];
  }
}

// ---------------- fused window attention: 512 threads, 2 heads in flight ----------------
__global__ __launch_bounds__(512) void winattn_kernel(
    const float* __restrict__ x, const float* __restrict__ mask,
    const bf16_t* __restrict__ wqkv, const float* __restrict__ qkv_b,
    const bf16_t* __restrict__ wproj, const float* __restrict__ proj_b,
    const float* __restrict__ biasws, float* __restrict__ out)
{
  extern __shared__ char smem[];
  bf16_t* xbf  = (bf16_t*)smem;                  // 64 x 520 token-major
  bf16_t* Obf  = xbf + NTOK * XS;                // 64 x 520 token-major
  bf16_t* grpb = Obf + NTOK * XS;                // 2 head-groups of bf16 buffers
  float*  scb  = (float*)(grpb + 2 * GRP_BF);    // 2 head-groups of scoresT

  const int t    = threadIdx.x;
  const int lane = t & 31;
  const int wv   = t >> 5;        // wave 0..15
  const int wl   = wv & 7;        // wave within group
  const int g    = wv >> 3;       // head-group 0/1
  const int win  = blockIdx.x;
  const float scale = 0.17677669529663687f;   // 32^-0.5

  bf16_t* q_s = grpb + g * GRP_BF;             // 64 x 40  [token][dh]
  bf16_t* k_s = q_s + NTOK * QKS;              // 64 x 40  [token][dh]
  bf16_t* vT  = k_s + NTOK * QKS;              // 32 x 72  [dh][token]
  bf16_t* wbf = vT + DHEAD * VTS;              // 64 x 72  [query][key]
  float*  sc  = scb + g * GRP_SC;              // 64 x 68  [key][query]

  // ---- stage x[win] into LDS as bf16 ----
  const float* xg = x + (size_t)win * NTOK * CDIM;
  for (int i = t; i < NTOK * CDIM / 4; i += 512) {
    v4f xv = *(const v4f*)(xg + i * 4);
    int row = (i * 4) / CDIM, col = (i * 4) % CDIM;
    bf16_t* d = xbf + row * XS + col;
    d[0] = (bf16_t)xv[0]; d[1] = (bf16_t)xv[1];
    d[2] = (bf16_t)xv[2]; d[3] = (bf16_t)xv[3];
  }
  __syncthreads();

  for (int h = 0; h < HEADS; h += 2) {
    const int h2 = h + g;

    // ---- qkv^T = W^T @ x^T : wave owns (token-tile nt, dh-half mtq) for q,k,v ----
    {
      const int nt  = wl & 3;      // token tile 0..3
      const int mtq = wl >> 2;     // dh half 0..1
      v8f cq = {}, ck = {}, cv = {};
      for (int kk = 0; kk < 16; ++kk) {
        v16bf b = load_B(xbf, XS, nt * 16, kk * 32, lane);        // shared across q,k,v
        const bf16_t* base = wqkv + (size_t)(kk * 96) * 512;
        v16bf aq = load_W(base + (size_t)(2 * h2 + mtq) * 512, lane);
        v16bf ak = load_W(base + (size_t)(32 + 2 * h2 + mtq) * 512, lane);
        v16bf av = load_W(base + (size_t)(64 + 2 * h2 + mtq) * 512, lane);
        cq = wmma_bf16(aq, b, cq);
        ck = wmma_bf16(ak, b, ck);
        cv = wmma_bf16(av, b, cv);
      }
      const int token = nt * 16 + (lane & 15);
      const int m0    = mtq * 16 + 8 * (lane >> 4);   // dh channel base within head
      v4f bq0 = *(const v4f*)(qkv_b + h2 * 32 + m0);
      v4f bq1 = *(const v4f*)(qkv_b + h2 * 32 + m0 + 4);
      v4f bk0 = *(const v4f*)(qkv_b + 512 + h2 * 32 + m0);
      v4f bk1 = *(const v4f*)(qkv_b + 512 + h2 * 32 + m0 + 4);
      v4f bv0 = *(const v4f*)(qkv_b + 1024 + h2 * 32 + m0);
      v4f bv1 = *(const v4f*)(qkv_b + 1024 + h2 * 32 + m0 + 4);
      union { v8bf v; bf16_t e[8]; } pq, pk;
#pragma unroll
      for (int r = 0; r < 8; ++r) {
        float bq = (r < 4) ? bq0[r] : bq1[r - 4];
        float bk = (r < 4) ? bk0[r] : bk1[r - 4];
        pq.e[r] = (bf16_t)((cq[r] + bq) * scale);
        pk.e[r] = (bf16_t)(ck[r] + bk);
      }
      *(v8bf*)(q_s + token * QKS + m0) = pq.v;        // contiguous 16B store
      *(v8bf*)(k_s + token * QKS + m0) = pk.v;        // contiguous 16B store
#pragma unroll
      for (int r = 0; r < 8; ++r) {
        float bv = (r < 4) ? bv0[r] : bv1[r - 4];
        vT[(m0 + r) * VTS + token] = (bf16_t)(cv[r] + bv);  // transposed (scattered)
      }
    }
    __syncthreads();

    // ---- S = q @ k^T : 2 tiles per wave, stored transposed scT[key][query] ----
    {
      const int tile0 = wl * 2, tile1 = wl * 2 + 1;
      const int mt0 = tile0 >> 2, nt0 = tile0 & 3;
      const int mt1 = tile1 >> 2, nt1 = tile1 & 3;
      v16bf a0 = load_A(q_s, QKS, mt0 * 16, 0, lane);
      v16bf b0 = load_B(k_s, QKS, nt0 * 16, 0, lane);
      v16bf a1 = load_A(q_s, QKS, mt1 * 16, 0, lane);
      v16bf b1 = load_B(k_s, QKS, nt1 * 16, 0, lane);
      v8f z = {};
      v8f c0 = wmma_bf16(a0, b0, z);
      v8f c1 = wmma_bf16(a1, b1, z);
      const int key0 = nt0 * 16 + (lane & 15), q00 = mt0 * 16 + 8 * (lane >> 4);
      const int key1 = nt1 * 16 + (lane & 15), q10 = mt1 * 16 + 8 * (lane >> 4);
      v4f s;
      s[0]=c0[0]; s[1]=c0[1]; s[2]=c0[2]; s[3]=c0[3];
      *(v4f*)(sc + key0 * SCS + q00) = s;
      s[0]=c0[4]; s[1]=c0[5]; s[2]=c0[6]; s[3]=c0[7];
      *(v4f*)(sc + key0 * SCS + q00 + 4) = s;
      s[0]=c1[0]; s[1]=c1[1]; s[2]=c1[2]; s[3]=c1[3];
      *(v4f*)(sc + key1 * SCS + q10) = s;
      s[0]=c1[4]; s[1]=c1[5]; s[2]=c1[6]; s[3]=c1[7];
      *(v4f*)(sc + key1 * SCS + q10 + 4) = s;
    }
    __syncthreads();

    // ---- softmax over keys: thread q owns column q (bank-conflict-free) ----
    if (t < 128) {
      const int sg = t >> 6, q = t & 63;
      float*  scs = scb + sg * GRP_SC;
      bf16_t* wbs = grpb + sg * GRP_BF + NTOK * QKS * 2 + DHEAD * VTS;
      const float* bh   = biasws + (h + sg) * 4096;   // [key][query]
      const float* mrow = mask + win * 64;
      float mx = -3.0e38f;
#pragma unroll
      for (int m = 0; m < 64; ++m) {
        float s = scs[m * SCS + q] + bh[m * 64 + q] + mrow[m];
        scs[m * SCS + q] = s;
        mx = fmaxf(mx, s);
      }
      float sum = 0.f;
#pragma unroll
      for (int m = 0; m < 64; ++m) sum += __expf(scs[m * SCS + q] - mx);
      const float inv = 1.0f / sum;
#pragma unroll
      for (int m = 0; m < 64; ++m)
        wbs[q * WBS + m] = (bf16_t)(__expf(scs[m * SCS + q] - mx) * inv);
    }
    __syncthreads();

    // ---- O^T = V^T @ W^T : one tile per wave, contiguous stores into Obf ----
    {
      const int mtd = wl >> 2;     // dh tile 0..1
      const int ntt = wl & 3;      // token tile 0..3
      v8f c = {};
#pragma unroll
      for (int kk = 0; kk < 2; ++kk) {
        v16bf a = load_A(vT, VTS, mtd * 16, kk * 32, lane);
        v16bf b = load_B(wbf, WBS, ntt * 16, kk * 32, lane);
        c = wmma_bf16(a, b, c);
      }
      const int token = ntt * 16 + (lane & 15);
      const int ch    = h2 * 32 + mtd * 16 + 8 * (lane >> 4);
      union { v8bf v; bf16_t e[8]; } p;
#pragma unroll
      for (int r = 0; r < 8; ++r) p.e[r] = (bf16_t)c[r];
      *(v8bf*)(Obf + token * XS + ch) = p.v;          // contiguous 16B store
    }
    __syncthreads();
  }

  // ---- out^T = Wp^T @ O^T : 8 tiles per wave, vectorized global stores ----
  float* og = out + (size_t)win * NTOK * CDIM;
  for (int j = 0; j < 8; ++j) {
    const int tile = wv + j * 16;
    const int mt = tile & 31;      // out-channel tile
    const int ntt = tile >> 5;     // token tile
    v8f c = {};
    for (int kk = 0; kk < 16; ++kk) {
      v16bf a = load_W(wproj + (size_t)(kk * 32 + mt) * 512, lane);
      v16bf b = load_B(Obf, XS, ntt * 16, kk * 32, lane);
      c = wmma_bf16(a, b, c);
    }
    const int token = ntt * 16 + (lane & 15);
    const int chb   = mt * 16 + 8 * (lane >> 4);
    v4f pb0 = *(const v4f*)(proj_b + chb);
    v4f pb1 = *(const v4f*)(proj_b + chb + 4);
    v4f s0, s1;
    s0[0]=c[0]+pb0[0]; s0[1]=c[1]+pb0[1]; s0[2]=c[2]+pb0[2]; s0[3]=c[3]+pb0[3];
    s1[0]=c[4]+pb1[0]; s1[1]=c[5]+pb1[1]; s1[2]=c[6]+pb1[2]; s1[3]=c[7]+pb1[3];
    float* p = og + (size_t)token * CDIM + chb;
    *(v4f*)p = s0;
    *(v4f*)(p + 4) = s1;
  }
}

// ---------------- launcher ----------------
extern "C" void kernel_launch(void* const* d_in, const int* in_sizes, int n_in,
                              void* d_out, int out_size, void* d_ws, size_t ws_size,
                              hipStream_t stream) {
  const float* x      = (const float*)d_in[0];
  const float* mask   = (const float*)d_in[1];
  const float* qkv_w  = (const float*)d_in[2];
  const float* qkv_b  = (const float*)d_in[3];
  const float* table  = (const float*)d_in[4];
  const float* proj_w = (const float*)d_in[5];
  const float* proj_b = (const float*)d_in[6];
  const int*   relidx = (const int*)d_in[7];
  float* out = (float*)d_out;

  bf16_t* wqkv   = (bf16_t*)((char*)d_ws + WS_QKVW_OFF);
  bf16_t* wproj  = (bf16_t*)((char*)d_ws + WS_PROJW_OFF);
  float*  biasws = (float*) ((char*)d_ws + WS_BIAS_OFF);

  prep_kernel<<<3072, 256, 0, stream>>>(qkv_w, proj_w, relidx, table, wqkv, wproj, biasws);

  const size_t lds =
      (size_t)NTOK * XS * 2 * 2 +        // xbf + Obf            = 133,120
      (size_t)2 * GRP_BF * 2 +           // 2 groups bf16 bufs   =  48,128
      (size_t)2 * GRP_SC * 4;            // 2 groups scoresT f32 =  34,816  => 216,064 B
  winattn_kernel<<<NWIN, 512, lds, stream>>>(x, mask, wqkv, qkv_b, wproj, proj_b, biasws, out);
}
